// MultiheadAttentionComp_26130581029589
// MI455X (gfx1250) — compile-verified, matrix operand order checked by reading, not run on hardware
//
#include <hip/hip_runtime.h>
#include <hip/hip_bf16.h>

// MI455X / gfx1250: wave32, WMMA bf16 16x16x32, async global->LDS copies.

typedef __attribute__((ext_vector_type(16))) __bf16 v16bf;
typedef __attribute__((ext_vector_type(8)))  float  v8f;
typedef __attribute__((ext_vector_type(4)))  int    v4i;

#define T_DIM   1024
#define B_DIM   4
#define E_DIM   1024
#define H_DIM   16
#define D_DIM   64
#define SIG_DIM 64
#define M_ROWS  (T_DIM * B_DIM)   // 4096

union FragBF {
    v16bf v;
    uint4 u[2];
};

__device__ __forceinline__ unsigned short f2bf(float f) {
    unsigned u = __builtin_bit_cast(unsigned, f);
    u += 0x7FFFu + ((u >> 16) & 1u);      // round-to-nearest-even
    return (unsigned short)(u >> 16);
}

// ---- async global->LDS (ASYNCcnt path), guarded with sync fallback --------
#if defined(__has_builtin)
#if __has_builtin(__builtin_amdgcn_global_load_async_to_lds_b128)
#define HAVE_ASYNC_LDS 1
#endif
#endif

#ifdef HAVE_ASYNC_LDS
// Signature (from compiler diagnostics): (int4 AS1* src, int4 AS3* dst, Ii, Ii)
#define ASYNC_COPY16(dst_lds, src_glb)                                         \
    __builtin_amdgcn_global_load_async_to_lds_b128(                            \
        (__attribute__((address_space(1))) v4i*)(src_glb),                     \
        (__attribute__((address_space(3))) v4i*)(dst_lds), 0, 0)
#if __has_builtin(__builtin_amdgcn_s_wait_asynccnt)
#define ASYNC_WAIT() __builtin_amdgcn_s_wait_asynccnt(0)
#else
#define ASYNC_WAIT() asm volatile("s_wait_asynccnt 0" ::: "memory")
#endif
#else
#define ASYNC_COPY16(dst_lds, src_glb)                                         \
    (*(uint4*)(dst_lds) = *(const uint4*)(src_glb))
#define ASYNC_WAIT() ((void)0)
#endif

// ---------------------------------------------------------------- convert
__global__ void cvt_f32_to_bf16(const float* __restrict__ in,
                                unsigned short* __restrict__ out, int n) {
    int i = blockIdx.x * blockDim.x + threadIdx.x;
    if (i < n) out[i] = f2bf(in[i]);
}

// ---------------------------------------------------------------- big GEMM
// C(M,N) = act((A(M,K) @ W(N,K)^T + bias) * scale)
// 256 threads = 8 waves; block tile 128x128; wave tile 32x64 (2x4 WMMA).
// Double-buffered LDS, async global->LDS copy of the next K-chunk overlapped
// with WMMA on the current one.
template <int OUT_BF16, int RELU>
__global__ void __launch_bounds__(256)
gemm_bias_wmma_128(const unsigned short* __restrict__ A,
                   const unsigned short* __restrict__ W,
                   const float* __restrict__ bias,
                   void* __restrict__ Cout,
                   int M, int N, int K, float scale) {
    __shared__ unsigned short As[2][128][32];
    __shared__ unsigned short Ws[2][128][32];

    const int tid  = threadIdx.x;
    const int lane = tid & 31;
    const int wave = tid >> 5;             // 0..7
    const int wm   = wave >> 1;            // 0..3 -> 32 rows each
    const int wn   = wave & 1;             // 0..1 -> 64 cols each
    const int m0   = blockIdx.y * 128;
    const int n0   = blockIdx.x * 128;

    v8f acc[2][4];
#pragma unroll
    for (int i = 0; i < 2; ++i)
#pragma unroll
        for (int j = 0; j < 4; ++j)
            acc[i][j] = (v8f){0.f, 0.f, 0.f, 0.f, 0.f, 0.f, 0.f, 0.f};

    const int ldRow = tid >> 1;            // 0..127
    const int ldCol = (tid & 1) * 16;      // 0 or 16
    const unsigned short* Aptr = &A[(size_t)(m0 + ldRow) * K + ldCol];
    const unsigned short* Wptr = &W[(size_t)(n0 + ldRow) * K + ldCol];

    const int kbase = (lane >> 4) << 3;    // 0 or 8 (ISA A/B frag K pattern)
    const int half  = lane & 15;

    // preload K-chunk 0 into buffer 0
    ASYNC_COPY16(&As[0][ldRow][ldCol],     Aptr);
    ASYNC_COPY16(&As[0][ldRow][ldCol + 8], Aptr + 8);
    ASYNC_COPY16(&Ws[0][ldRow][ldCol],     Wptr);
    ASYNC_COPY16(&Ws[0][ldRow][ldCol + 8], Wptr + 8);

    int buf = 0;
    for (int k0 = 0; k0 < K; k0 += 32) {
        ASYNC_WAIT();
        __syncthreads();                   // tile(k0) resident in As/Ws[buf]

        if (k0 + 32 < K) {                 // overlap next copy with compute
            const int nb = buf ^ 1;
            ASYNC_COPY16(&As[nb][ldRow][ldCol],     Aptr + k0 + 32);
            ASYNC_COPY16(&As[nb][ldRow][ldCol + 8], Aptr + k0 + 40);
            ASYNC_COPY16(&Ws[nb][ldRow][ldCol],     Wptr + k0 + 32);
            ASYNC_COPY16(&Ws[nb][ldRow][ldCol + 8], Wptr + k0 + 40);
        }

        FragBF a[2], bq[4];
#pragma unroll
        for (int mt = 0; mt < 2; ++mt) {
            const int row = wm * 32 + mt * 16 + half;
            a[mt].u[0] = *(const uint4*)&As[buf][row][kbase];
            a[mt].u[1] = *(const uint4*)&As[buf][row][kbase + 16];
        }
#pragma unroll
        for (int nt = 0; nt < 4; ++nt) {
            const int row = wn * 64 + nt * 16 + half;
            bq[nt].u[0] = *(const uint4*)&Ws[buf][row][kbase];
            bq[nt].u[1] = *(const uint4*)&Ws[buf][row][kbase + 16];
        }
#pragma unroll
        for (int mt = 0; mt < 2; ++mt)
#pragma unroll
            for (int nt = 0; nt < 4; ++nt)
                acc[mt][nt] = __builtin_amdgcn_wmma_f32_16x16x32_bf16(
                    false, a[mt].v, false, bq[nt].v, (short)0, acc[mt][nt], false, false);
        buf ^= 1;
    }

#pragma unroll
    for (int nt = 0; nt < 4; ++nt) {
        const int gn = n0 + wn * 64 + nt * 16 + half;
        const float bv = bias[gn];
#pragma unroll
        for (int mt = 0; mt < 2; ++mt)
#pragma unroll
            for (int r = 0; r < 8; ++r) {
                const int mi = ((lane >> 4) << 3) + r;   // C layout: M = r (+8 hi half)
                const int gm = m0 + wm * 32 + mt * 16 + mi;
                float v = (acc[mt][nt][r] + bv) * scale;
                if (RELU) v = v > 0.f ? v : 0.f;
                if (OUT_BF16)
                    ((unsigned short*)Cout)[(size_t)gm * N + gn] = f2bf(v);
                else
                    ((float*)Cout)[(size_t)gm * N + gn] = v;
            }
    }
}

// ---------------------------------------------------------------- small GEMM
// 64x64 tile version for the N=64 signature projection (t2 = t1 @ W2^T + b2).
template <int OUT_BF16, int RELU>
__global__ void __launch_bounds__(128)
gemm_bias_wmma_64(const unsigned short* __restrict__ A,
                  const unsigned short* __restrict__ W,
                  const float* __restrict__ bias,
                  void* __restrict__ Cout,
                  int M, int N, int K, float scale) {
    __shared__ unsigned short As[64][32];
    __shared__ unsigned short Ws[64][32];

    const int tid  = threadIdx.x;
    const int lane = tid & 31;
    const int wave = tid >> 5;
    const int wm   = wave >> 1;
    const int wn   = wave & 1;
    const int m0   = blockIdx.y * 64;
    const int n0   = blockIdx.x * 64;

    v8f acc[2][2];
#pragma unroll
    for (int i = 0; i < 2; ++i)
#pragma unroll
        for (int j = 0; j < 2; ++j)
            acc[i][j] = (v8f){0.f, 0.f, 0.f, 0.f, 0.f, 0.f, 0.f, 0.f};

    const int ldRow = tid >> 1;
    const int ldCol = (tid & 1) * 16;
    const int kbase = (lane >> 4) << 3;
    const int half  = lane & 15;

    for (int k0 = 0; k0 < K; k0 += 32) {
        if (k0 + 32 < K) {
            __builtin_prefetch(&A[(size_t)(m0 + ldRow) * K + k0 + 32 + ldCol], 0, 0);
            __builtin_prefetch(&W[(size_t)(n0 + ldRow) * K + k0 + 32 + ldCol], 0, 0);
        }
        *(uint4*)&As[ldRow][ldCol]     = *(const uint4*)&A[(size_t)(m0 + ldRow) * K + k0 + ldCol];
        *(uint4*)&As[ldRow][ldCol + 8] = *(const uint4*)&A[(size_t)(m0 + ldRow) * K + k0 + ldCol + 8];
        *(uint4*)&Ws[ldRow][ldCol]     = *(const uint4*)&W[(size_t)(n0 + ldRow) * K + k0 + ldCol];
        *(uint4*)&Ws[ldRow][ldCol + 8] = *(const uint4*)&W[(size_t)(n0 + ldRow) * K + k0 + ldCol + 8];
        __syncthreads();

        FragBF a[2], bf[2];
#pragma unroll
        for (int mt = 0; mt < 2; ++mt) {
            const int row = wm * 32 + mt * 16 + half;
            a[mt].u[0] = *(const uint4*)&As[row][kbase];
            a[mt].u[1] = *(const uint4*)&As[row][kbase + 16];
        }
#pragma unroll
        for (int nt = 0; nt < 2; ++nt) {
            const int row = wn * 32 + nt * 16 + half;
            bf[nt].u[0] = *(const uint4*)&Ws[row][kbase];
            bf[nt].u[1] = *(const uint4*)&Ws[row][kbase + 16];
        }
#pragma unroll
        for (int mt = 0; mt < 2; ++mt)
#pragma unroll
            for (int nt = 0; nt < 2; ++nt)
                acc[mt][nt] = __builtin_amdgcn_wmma_f32_16x16x32_bf16(
                    false, a[mt].v, false, bf[nt].v, (short)0, acc[mt][nt], false, false);
        __syncthreads();
    }

#pragma unroll
    for (int nt = 0; nt < 2; ++nt) {
        const int gn = n0 + wn * 32 + nt * 16 + half;
        const float bv = bias[gn];
#pragma unroll
        for (int mt = 0; mt < 2; ++mt)
#pragma unroll
            for (int r = 0; r < 8; ++r) {
                const int mi = ((lane >> 4) << 3) + r;
                const int gm = m0 + wm * 32 + mt * 16 + mi;
                float v = (acc[mt][nt][r] + bv) * scale;
                if (RELU) v = v > 0.f ? v : 0.f;
                if (OUT_BF16)
                    ((unsigned short*)Cout)[(size_t)gm * N + gn] = f2bf(v);
                else
                    ((float*)Cout)[(size_t)gm * N + gn] = v;
            }
    }
}

// ---------------------------------------------------------------- gate
__global__ void gate_softmax(const float* __restrict__ t2,
                             const float* __restrict__ sig,
                             float* __restrict__ gate) {
    int row = blockIdx.x * blockDim.x + threadIdx.x;
    if (row >= M_ROWS) return;
    float lg[H_DIM];
    float mx = -3.0e38f;
    for (int h = 0; h < H_DIM; ++h) {
        float s = 0.f;
        for (int j = 0; j < SIG_DIM; ++j)
            s += t2[(size_t)row * SIG_DIM + j] * sig[h * SIG_DIM + j];
        lg[h] = s;
        mx = fmaxf(mx, s);
    }
    float d = 0.f;
    for (int h = 0; h < H_DIM; ++h) { lg[h] = __expf(lg[h] - mx); d += lg[h]; }
    float inv = 1.f / d;
    for (int h = 0; h < H_DIM; ++h) gate[(size_t)row * H_DIM + h] = lg[h] * inv;
}

// ---------------------------------------------------------------- flash attention
// One block = one (b,h) x one 64-row Q tile. 4 waves; wave owns 16 Q rows.
// Q and K tiles staged via async global->LDS; V transposed through VGPRs.
__global__ void __launch_bounds__(128)
attn_flash_wmma(const unsigned short* __restrict__ qbuf,
                const unsigned short* __restrict__ kbuf,
                const unsigned short* __restrict__ vbuf,
                const float* __restrict__ gate,
                unsigned short* __restrict__ og) {
    __shared__ unsigned short Qs[64][64];
    __shared__ unsigned short Ks[32][64];
    __shared__ unsigned short Vst[64][40];   // [d][key], padded stride
    __shared__ unsigned short Ps[64][32];
    __shared__ float Sb[64][32];
    __shared__ float mbuf[64], lbuf[64], abuf[64];

    const int tid  = threadIdx.x;
    const int lane = tid & 31;
    const int wave = tid >> 5;
    const int bh   = blockIdx.y;            // 0..63
    const int b    = bh >> 4;
    const int h    = bh & 15;
    const int q0   = blockIdx.x * 64;

    for (int idx = tid; idx < 512; idx += 128) {   // 64x64 Q tile, async
        int row = idx >> 3, dg = (idx & 7) * 8;
        ASYNC_COPY16(&Qs[row][dg],
                     &qbuf[((size_t)(q0 + row) * B_DIM + b) * E_DIM + h * D_DIM + dg]);
    }
    if (tid < 64) { mbuf[tid] = -3.0e38f; lbuf[tid] = 0.f; }

    v8f accO[4];
#pragma unroll
    for (int i = 0; i < 4; ++i)
        accO[i] = (v8f){0.f, 0.f, 0.f, 0.f, 0.f, 0.f, 0.f, 0.f};

    ASYNC_WAIT();
    __syncthreads();

    const int kbase = (lane >> 4) << 3;
    const int half  = lane & 15;

    for (int s0 = 0; s0 < T_DIM; s0 += 32) {
        for (int idx = tid; idx < 256; idx += 128) {   // 32x64 K and V tiles
            int key = idx >> 3, dg = (idx & 7) * 8;
            size_t base = ((size_t)(s0 + key) * B_DIM + b) * E_DIM + h * D_DIM + dg;
            ASYNC_COPY16(&Ks[key][dg], &kbuf[base]);   // K: straight async copy
            uint4 vv = *(const uint4*)&vbuf[base];     // V: transpose via VGPRs
            const unsigned short* vs = (const unsigned short*)&vv;
#pragma unroll
            for (int j = 0; j < 8; ++j) Vst[dg + j][key] = vs[j];
        }
        ASYNC_WAIT();
        __syncthreads();

        // S(16x32 per wave) = Q(16x64) @ K^T : chained K=32 WMMAs
#pragma unroll
        for (int nt = 0; nt < 2; ++nt) {
            v8f s = (v8f){0.f, 0.f, 0.f, 0.f, 0.f, 0.f, 0.f, 0.f};
#pragma unroll
            for (int kc = 0; kc < 2; ++kc) {
                FragBF a, bb;
                const int qrow = wave * 16 + half;
                a.u[0] = *(const uint4*)&Qs[qrow][kc * 32 + kbase];
                a.u[1] = *(const uint4*)&Qs[qrow][kc * 32 + kbase + 16];
                const int krow = nt * 16 + half;
                bb.u[0] = *(const uint4*)&Ks[krow][kc * 32 + kbase];
                bb.u[1] = *(const uint4*)&Ks[krow][kc * 32 + kbase + 16];
                s = __builtin_amdgcn_wmma_f32_16x16x32_bf16(
                    false, a.v, false, bb.v, (short)0, s, false, false);
            }
#pragma unroll
            for (int r = 0; r < 8; ++r) {
                int mi = ((lane >> 4) << 3) + r;
                Sb[wave * 16 + mi][nt * 16 + half] = s[r];
            }
        }
        __syncthreads();

        // online softmax, one thread per row (q pre-scaled by d^-0.5)
        if (tid < 64) {
            float mold = mbuf[tid], tm = mold;
#pragma unroll
            for (int j = 0; j < 32; ++j) tm = fmaxf(tm, Sb[tid][j]);
            float alpha = __expf(mold - tm);
            float lsum = 0.f;
#pragma unroll
            for (int j = 0; j < 32; ++j) {
                float p = __expf(Sb[tid][j] - tm);
                Ps[tid][j] = f2bf(p);
                lsum += p;
            }
            mbuf[tid] = tm;
            lbuf[tid] = lbuf[tid] * alpha + lsum;
            abuf[tid] = alpha;
        }
        __syncthreads();

        // O = O*alpha + P(16x32) @ V(32x64)
        FragBF p;
        {
            const int prow = wave * 16 + half;
            p.u[0] = *(const uint4*)&Ps[prow][kbase];
            p.u[1] = *(const uint4*)&Ps[prow][kbase + 16];
        }
#pragma unroll
        for (int nt = 0; nt < 4; ++nt) {
#pragma unroll
            for (int r = 0; r < 8; ++r) {
                int mi = ((lane >> 4) << 3) + r;
                accO[nt][r] *= abuf[wave * 16 + mi];
            }
            FragBF vv;
            vv.u[0] = *(const uint4*)&Vst[nt * 16 + half][kbase];
            vv.u[1] = *(const uint4*)&Vst[nt * 16 + half][kbase + 16];
            accO[nt] = __builtin_amdgcn_wmma_f32_16x16x32_bf16(
                false, p.v, false, vv.v, (short)0, accO[nt], false, false);
        }
        __syncthreads();
    }

    // (O / l) * gate -> bf16 (T*B, E)
#pragma unroll
    for (int nt = 0; nt < 4; ++nt)
#pragma unroll
        for (int r = 0; r < 8; ++r) {
            int mi   = ((lane >> 4) << 3) + r;
            int trow = q0 + wave * 16 + mi;
            float g  = gate[(size_t)(trow * B_DIM + b) * H_DIM + h];
            float v  = accO[nt][r] / lbuf[wave * 16 + mi] * g;
            og[((size_t)trow * B_DIM + b) * E_DIM + h * D_DIM + nt * 16 + half] = f2bf(v);
        }
}

// ---------------------------------------------------------------- launch
extern "C" void kernel_launch(void* const* d_in, const int* in_sizes, int n_in,
                              void* d_out, int out_size, void* d_ws, size_t ws_size,
                              hipStream_t stream) {
    const float* x   = (const float*)d_in[0];
    const float* Wq  = (const float*)d_in[1];
    const float* bq  = (const float*)d_in[2];
    const float* Wk  = (const float*)d_in[3];
    const float* bk  = (const float*)d_in[4];
    const float* Wv  = (const float*)d_in[5];
    const float* bv  = (const float*)d_in[6];
    const float* Wo  = (const float*)d_in[7];
    const float* bo  = (const float*)d_in[8];
    const float* W1  = (const float*)d_in[9];
    const float* b1  = (const float*)d_in[10];
    const float* W2  = (const float*)d_in[11];
    const float* b2  = (const float*)d_in[12];
    const float* sig = (const float*)d_in[13];

    char* ws = (char*)d_ws;
    size_t off = 0;
    auto alloc = [&](size_t bytes) -> void* {
        void* p = ws + off;
        off += (bytes + 255) & ~(size_t)255;
        return p;
    };
    unsigned short* xb  = (unsigned short*)alloc((size_t)M_ROWS * E_DIM * 2);
    unsigned short* wqb = (unsigned short*)alloc((size_t)E_DIM * E_DIM * 2);
    unsigned short* wkb = (unsigned short*)alloc((size_t)E_DIM * E_DIM * 2);
    unsigned short* wvb = (unsigned short*)alloc((size_t)E_DIM * E_DIM * 2);
    unsigned short* wob = (unsigned short*)alloc((size_t)E_DIM * E_DIM * 2);
    unsigned short* w1b = (unsigned short*)alloc((size_t)E_DIM * E_DIM * 2);
    unsigned short* w2b = (unsigned short*)alloc((size_t)SIG_DIM * E_DIM * 2);
    unsigned short* qb  = (unsigned short*)alloc((size_t)M_ROWS * E_DIM * 2);
    unsigned short* kb  = (unsigned short*)alloc((size_t)M_ROWS * E_DIM * 2);
    unsigned short* vb  = (unsigned short*)alloc((size_t)M_ROWS * E_DIM * 2);
    unsigned short* t1b = (unsigned short*)alloc((size_t)M_ROWS * E_DIM * 2);
    float*          t2f = (float*)alloc((size_t)M_ROWS * SIG_DIM * 4);
    float*          gtf = (float*)alloc((size_t)M_ROWS * H_DIM * 4);
    unsigned short* ogb = (unsigned short*)alloc((size_t)M_ROWS * E_DIM * 2);
    (void)ws_size; (void)n_in; (void)in_sizes; (void)out_size;

    auto cvt = [&](const float* src, unsigned short* dst, int n) {
        cvt_f32_to_bf16<<<(n + 255) / 256, 256, 0, stream>>>(src, dst, n);
    };
    cvt(x,  xb,  M_ROWS * E_DIM);
    cvt(Wq, wqb, E_DIM * E_DIM);
    cvt(Wk, wkb, E_DIM * E_DIM);
    cvt(Wv, wvb, E_DIM * E_DIM);
    cvt(Wo, wob, E_DIM * E_DIM);
    cvt(W1, w1b, E_DIM * E_DIM);
    cvt(W2, w2b, SIG_DIM * E_DIM);

    const dim3 blk256(256), blk128(128);
    const dim3 gBig(E_DIM / 128, M_ROWS / 128);      // 8 x 32 blocks
    const dim3 gSig(SIG_DIM / 64, M_ROWS / 64);      // 1 x 64 blocks

    // q (scaled), k, v — bf16 out
    gemm_bias_wmma_128<1, 0><<<gBig, blk256, 0, stream>>>(xb, wqb, bq, qb, M_ROWS, E_DIM, E_DIM, 0.125f);
    gemm_bias_wmma_128<1, 0><<<gBig, blk256, 0, stream>>>(xb, wkb, bk, kb, M_ROWS, E_DIM, E_DIM, 1.0f);
    gemm_bias_wmma_128<1, 0><<<gBig, blk256, 0, stream>>>(xb, wvb, bv, vb, M_ROWS, E_DIM, E_DIM, 1.0f);
    // gate MLP
    gemm_bias_wmma_128<1, 1><<<gBig, blk256, 0, stream>>>(xb, w1b, b1, t1b, M_ROWS, E_DIM, E_DIM, 1.0f);
    gemm_bias_wmma_64<0, 0><<<gSig, blk128, 0, stream>>>(t1b, w2b, b2, t2f, M_ROWS, SIG_DIM, E_DIM, 1.0f);
    gate_softmax<<<M_ROWS / 256, 256, 0, stream>>>(t2f, sig, gtf);
    // flash attention + gating -> og bf16
    attn_flash_wmma<<<dim3(T_DIM / 64, B_DIM * H_DIM), blk128, 0, stream>>>(qb, kb, vb, gtf, ogb);
    // output projection (fp32 out)
    gemm_bias_wmma_128<0, 0><<<gBig, blk256, 0, stream>>>(ogb, wob, bo, (float*)d_out,
                                                          M_ROWS, E_DIM, E_DIM, 1.0f);
}
